// GPT2PseudoAttention_13194139533904
// MI455X (gfx1250) — matrix operand order repaired and checked by reading, not compiled, not run on hardware
//
#include <hip/hip_runtime.h>
#include <hip/hip_bf16.h>

// ---------------- types ----------------
typedef _Float16 v8h  __attribute__((ext_vector_type(8)));
typedef _Float16 v16h __attribute__((ext_vector_type(16)));
typedef float    v8f  __attribute__((ext_vector_type(8)));
typedef unsigned int u32x4 __attribute__((ext_vector_type(4)));
typedef int          i32x4 __attribute__((ext_vector_type(4)));
typedef int          i32x8 __attribute__((ext_vector_type(8)));

#define EMBED 1024
#define HEADS 16
#define HDIM  64
#define SEQ   1023
#define BATCH 4
#define MROWS (BATCH*SEQ)   // 4092
#define MPAD  4096
#define NQKV  3072
#define KEYS  (SEQ+1)       // 1024 (key 0 = image token)

// LDS tile geometry shared by TDM descriptor and compute side:
// 64 rows x 32 halves, padded to 40-half row stride (TDM pad: every 16 DWORDs add 4 DWORDs)
#define TILE_ROWS   64
#define TILE_K      32
#define LDS_STRIDE  40                       // halves
#define TILE_BYTES  (TILE_ROWS * LDS_STRIDE * 2)   // 5120 B
#define GEMM_LDS_BYTES (2 * TILE_BYTES)            // As + Bs

#if defined(__gfx1250__) && __has_builtin(__builtin_amdgcn_tensor_load_to_lds)
#define USE_TDM 1
#else
#define USE_TDM 0
#endif

__device__ __forceinline__ v16h cat8(v8h lo, v8h hi) {
  return __builtin_shufflevector(lo, hi, 0,1,2,3,4,5,6,7,8,9,10,11,12,13,14,15);
}

#if USE_TDM
// Issue one TDM 2-D tile load: 64 rows x 32 f16, row stride = stride_elems, into
// LDS at lds_off with padded row stride 40 halves (pad 4 DWORDs every 16 DWORDs).
__device__ __forceinline__ void tdm_load_tile(unsigned lds_off, const _Float16* gptr,
                                              unsigned stride_elems) {
  unsigned long long ga = (unsigned long long)gptr;
  u32x4 g0;
  g0[0] = 1u;                                    // count=1, user descriptor
  g0[1] = lds_off;                               // LDS byte address
  g0[2] = (unsigned)(ga & 0xffffffffu);          // global_addr[31:0]
  g0[3] = (unsigned)((ga >> 32) & 0x1ffffffu)    // global_addr[56:32]
        | (2u << 30);                            // type=2 ("image")
  i32x8 g1;
  g1[0] = (1 << 16)                              // data_size = 2 bytes
        | (1 << 20)                              // pad_enable
        | (3 << 22)                              // pad_interval: 16 DWORDs (64B = one row)
        | (3 << 25);                             // pad_amount: 4 DWORDs (16B = 8 halves)
  g1[1] = (TILE_K << 16);                        // tensor_dim0[15:0] = 32
  g1[2] = (TILE_ROWS << 16);                     // tensor_dim0[31:16]=0, tensor_dim1[15:0]=64
  g1[3] = (TILE_K << 16);                        // tensor_dim1[31:16]=0, tile_dim0 = 32
  g1[4] = TILE_ROWS;                             // tile_dim1 = 64, tile_dim2 = 0 (2-D)
  g1[5] = (int)stride_elems;                     // tensor_dim0_stride[31:0]
  g1[6] = 0;                                     // stride hi / dim1_stride (unused, 2-D)
  g1[7] = 0;
  i32x4 gz4 = {0, 0, 0, 0};                      // groups 2/3 unused (tile_dim3/4 = 0)
  i32x8 gz8 = {0, 0, 0, 0, 0, 0, 0, 0};          // extra group (clang-23 6-arg form), unused
  __builtin_amdgcn_tensor_load_to_lds(g0, g1, gz4, gz4, gz8, 0);
}
#endif

// ---------------- fp32 -> f16 convert of X (padded to MPAD rows) ----------------
__global__ void cvt_x_kernel(const float* __restrict__ x, _Float16* __restrict__ xh) {
  int idx = blockIdx.x * blockDim.x + threadIdx.x;
  if (idx >= MPAD * EMBED) return;
  int m = idx >> 10;
  float v = 0.f;
  if (m < MROWS) v = x[idx];
  xh[idx] = (_Float16)v;
}

// wt[n][k] = (f16) w[k][n]   (w is [K][N] row-major fp32)
__global__ void transpose_f16_kernel(const float* __restrict__ w, _Float16* __restrict__ wt,
                                     int K, int N) {
  int idx = blockIdx.x * blockDim.x + threadIdx.x;
  if (idx >= K * N) return;
  int n = idx / K, k = idx - n * K;
  wt[idx] = (_Float16)w[(long)k * N + n];
}

// ---------------- image K/V projections (tiny: 4x1024x1024 x2) ----------------
__global__ void img_kv_kernel(const float* __restrict__ img,
                              const float* __restrict__ ukw, const float* __restrict__ ukb,
                              const float* __restrict__ uvw, const float* __restrict__ uvb,
                              _Float16* __restrict__ Kb, _Float16* __restrict__ Vb) {
  int idx = blockIdx.x * blockDim.x + threadIdx.x;   // [0, 2*B*E)
  if (idx >= 2 * BATCH * EMBED) return;
  int which = (idx >= BATCH * EMBED) ? 1 : 0;
  int r = idx - which * BATCH * EMBED;
  int b = r >> 10, n = r & (EMBED - 1);
  const float* w  = which ? uvw : ukw;               // [n][k] orientation (img @ W^T)
  const float* bs = which ? uvb : ukb;
  const float* xi = img + (long)b * EMBED;
  const float* wr = w + (long)n * EMBED;
  float acc = bs[n];
#pragma unroll 4
  for (int k = 0; k < EMBED; k += 4)
    acc += xi[k]*wr[k] + xi[k+1]*wr[k+1] + xi[k+2]*wr[k+2] + xi[k+3]*wr[k+3];
  int h = n >> 6, d = n & 63;
  _Float16* dst = which ? Vb : Kb;
  dst[(((long)b * HEADS + h) * KEYS + 0) * HDIM + d] = (_Float16)acc;  // key slot 0 = image
}

// ---------------- WMMA GEMM: C[M x N] = A[MPAD x K](f16) * BT[N x K]^T + bias ----------------
// Tiles staged global->LDS by the Tensor Data Mover (TENSORcnt), WMMA f16->f32 compute.
// MODE 0: scatter f16 into Q/K/V head-major buffers (QKV projection)
// MODE 1: fp32 output + bias (output projection)
template<int MODE>
__global__ __launch_bounds__(128)
void gemm_wmma_kernel(const _Float16* __restrict__ A, const _Float16* __restrict__ BT,
                      const float* __restrict__ bias, int K, int N,
                      _Float16* __restrict__ Qb, _Float16* __restrict__ Kb,
                      _Float16* __restrict__ Vb, float* __restrict__ Cout) {
  extern __shared__ _Float16 smem[];               // dynamic LDS: offsets known for D#
  _Float16* As = smem;                             // LDS byte offset 0
  _Float16* Bs = smem + TILE_ROWS * LDS_STRIDE;    // LDS byte offset TILE_BYTES
  const int m0 = blockIdx.y * 64;
  const int n0 = blockIdx.x * 64;
  const int tid  = threadIdx.x;
  const int lane = tid & 31, wid = tid >> 5;
  const int lg = lane >> 4, ln = lane & 15;        // half-wave / index-in-half
  const int waveM = wid >> 1, waveN = wid & 1;     // 2x2 wave grid, 32x32 per wave

  v8f acc[2][2] = {};                              // [ms][ns] 16x16 f32 accumulators

#if !USE_TDM
  const int lrow = tid >> 1;                       // fallback staging split
  const int lseg = (tid & 1) * 16;
#endif

  for (int k0 = 0; k0 < K; k0 += TILE_K) {
    __syncthreads();                               // previous compute done before overwrite
#if USE_TDM
    if (wid == 0) {                                // one wave drives the DMA engine
      tdm_load_tile(0,          A  + (long)m0 * K + k0, (unsigned)K);
      tdm_load_tile(TILE_BYTES, BT + (long)n0 * K + k0, (unsigned)K);
      __builtin_amdgcn_s_wait_tensorcnt(0);
    }
#else
    {
      const _Float16* ag = A  + (long)(m0 + lrow) * K + k0 + lseg;
      const _Float16* bg = BT + (long)(n0 + lrow) * K + k0 + lseg;
      v8h a0 = *(const v8h*)(ag);
      v8h a1 = *(const v8h*)(ag + 8);
      v8h b0 = *(const v8h*)(bg);
      v8h b1 = *(const v8h*)(bg + 8);
      *(v8h*)(&As[lrow * LDS_STRIDE + lseg])     = a0;
      *(v8h*)(&As[lrow * LDS_STRIDE + lseg + 8]) = a1;
      *(v8h*)(&Bs[lrow * LDS_STRIDE + lseg])     = b0;
      *(v8h*)(&Bs[lrow * LDS_STRIDE + lseg + 8]) = b1;
    }
#endif
    if (k0 + TILE_K < K) {                         // warm L2 for the next k-tile
      __builtin_prefetch(A  + (long)(m0 + (tid >> 1)) * K + k0 + TILE_K, 0, 0);
      __builtin_prefetch(BT + (long)(n0 + (tid >> 1)) * K + k0 + TILE_K, 0, 0);
    }
    __syncthreads();

    // A fragment (16-bit A 16x32): halves0-7 -> k=g*8+h ; halves8-15 -> k=16+g*8+h
    v16h af[2], bf[2];
#pragma unroll
    for (int ms = 0; ms < 2; ++ms) {
      int mr = waveM * 32 + ms * 16 + ln;
      v8h lo = *(const v8h*)(&As[mr * LDS_STRIDE + lg * 8]);
      v8h hi = *(const v8h*)(&As[mr * LDS_STRIDE + 16 + lg * 8]);
      af[ms] = cat8(lo, hi);
    }
    // B fragment (16-bit B 32x16): lane n = ln, halves0-7 -> k=g*8+h ; halves8-15 -> k=16+g*8+h
#pragma unroll
    for (int ns = 0; ns < 2; ++ns) {
      int nr = waveN * 32 + ns * 16 + ln;
      v8h lo = *(const v8h*)(&Bs[nr * LDS_STRIDE + lg * 8]);
      v8h hi = *(const v8h*)(&Bs[nr * LDS_STRIDE + 16 + lg * 8]);
      bf[ns] = cat8(lo, hi);
    }
#pragma unroll
    for (int ms = 0; ms < 2; ++ms)
#pragma unroll
      for (int ns = 0; ns < 2; ++ns)
        acc[ms][ns] = __builtin_amdgcn_wmma_f32_16x16x32_f16(
            false, af[ms], false, bf[ns], (short)0, acc[ms][ns], false, false);
  }

  // ---- epilogue ----
#pragma unroll
  for (int ms = 0; ms < 2; ++ms)
#pragma unroll
    for (int ns = 0; ns < 2; ++ns) {
      int gn = n0 + waveN * 32 + ns * 16 + ln;
      float bv = bias[gn];
#pragma unroll
      for (int r = 0; r < 8; ++r) {
        int gm = m0 + waveM * 32 + ms * 16 + r + 8 * lg;
        if (gm >= MROWS) continue;
        float val = acc[ms][ns][r] + bv;
        if (MODE == 0) {
          int b = gm / SEQ, s = gm - b * SEQ;
          int d = gn & 63;
          if (gn < EMBED) {                 // Q, pre-scaled by 1/sqrt(HDIM)
            int h = gn >> 6;
            Qb[(((long)b * HEADS + h) * 1024 + s) * HDIM + d] = (_Float16)(val * 0.125f);
          } else if (gn < 2 * EMBED) {      // K words -> key slot s+1
            int h = (gn - EMBED) >> 6;
            Kb[(((long)b * HEADS + h) * KEYS + (s + 1)) * HDIM + d] = (_Float16)val;
          } else {                          // V words -> key slot s+1
            int h = (gn - 2 * EMBED) >> 6;
            Vb[(((long)b * HEADS + h) * KEYS + (s + 1)) * HDIM + d] = (_Float16)val;
          }
        } else {
          Cout[(long)gm * N + gn] = val;
        }
      }
    }
}

// ---------------- flash attention: one wave per (b, h, 16-query tile) ----------------
__global__ __launch_bounds__(32)
void attn_kernel(const _Float16* __restrict__ Qb, const _Float16* __restrict__ Kb,
                 const _Float16* __restrict__ Vb, const float* __restrict__ amask,
                 _Float16* __restrict__ AttOut) {
  __shared__ __align__(16) _Float16 Pt[16 * 40];   // probabilities tile [m][kcol]
  __shared__ __align__(16) _Float16 Vt[64 * 40];   // V transposed [d][kcol]
  const int bid = blockIdx.x;                      // b*H*64 + h*64 + qt
  const int qt = bid & 63;
  const int h  = (bid >> 6) & 15;
  const int b  = bid >> 10;
  const int lane = threadIdx.x;
  const int lg = lane >> 4, ln = lane & 15;
  const int qbase = qt * 16;

  const _Float16* Qrow = Qb + ((long)b * HEADS + h) * 1024 * HDIM;
  const _Float16* Krow = Kb + ((long)b * HEADS + h) * KEYS * HDIM;
  const _Float16* Vrow = Vb + ((long)b * HEADS + h) * KEYS * HDIM;

  // Q fragments: A 16x32 layout, two d-chunks of 32
  v16h qf[2];
  {
    const _Float16* qp = Qrow + (long)(qbase + ln) * HDIM;
#pragma unroll
    for (int c = 0; c < 2; ++c) {
      v8h lo = *(const v8h*)(qp + c * 32 + lg * 8);
      v8h hi = *(const v8h*)(qp + c * 32 + 16 + lg * 8);
      qf[c] = cat8(lo, hi);
    }
  }

  float rmax[8], rsum[8];
  v8f o[4] = {};
#pragma unroll
  for (int r = 0; r < 8; ++r) { rmax[r] = -1e30f; rsum[r] = 0.f; }

  int kend = qbase + 17; if (kend > KEYS) kend = KEYS;   // causal: kj <= qi+1
  const int ktiles = (kend + 31) >> 5;

  for (int kt = 0; kt < ktiles; ++kt) {
    const int kb = kt * 32;

    // stage V tile transposed: Vt[d][key-local]
    {
      const _Float16* vp = Vrow + (long)(kb + lane) * HDIM;
#pragma unroll
      for (int dd = 0; dd < 64; dd += 8) {
        v8h v = *(const v8h*)(vp + dd);
#pragma unroll
        for (int e = 0; e < 8; ++e) Vt[(dd + e) * 40 + lane] = v[e];
      }
    }

    // scores S = Q * K^T  (contraction over d=64 -> two chained WMMAs per 16-key subtile)
    v8f sc[2];
#pragma unroll
    for (int nt = 0; nt < 2; ++nt) {
      v8f z = {};
#pragma unroll
      for (int c = 0; c < 2; ++c) {
        const _Float16* kp = Krow + (long)(kb + nt * 16 + ln) * HDIM + c * 32 + lg * 8;
        v8h lo = *(const v8h*)(kp);
        v8h hi = *(const v8h*)(kp + 16);
        z = __builtin_amdgcn_wmma_f32_16x16x32_f16(false, qf[c], false, cat8(lo, hi),
                                                   (short)0, z, false, false);
      }
      sc[nt] = z;
    }

    // causal mask + attention_mask + online softmax (row lives in one half-wave)
    float mval[2];
#pragma unroll
    for (int nt = 0; nt < 2; ++nt) {
      int kj = kb + nt * 16 + ln;
      mval[nt] = (kj == 0) ? 0.f : amask[(long)b * SEQ + (kj - 1)];
    }
    float factor[8];
#pragma unroll
    for (int r = 0; r < 8; ++r) {
      int qi = qbase + r + 8 * lg;
      int kj0 = kb + ln, kj1 = kb + 16 + ln;
      float s0 = ((kj0 <= qi + 1) ? sc[0][r] : -10000.f) + mval[0];
      float s1 = ((kj1 <= qi + 1) ? sc[1][r] : -10000.f) + mval[1];
      float mx = fmaxf(s0, s1);
#pragma unroll
      for (int off = 1; off < 16; off <<= 1) mx = fmaxf(mx, __shfl_xor(mx, off, 16));
      float nm = fmaxf(rmax[r], mx);
      float f  = __expf(rmax[r] - nm);
      float p0 = __expf(s0 - nm), p1 = __expf(s1 - nm);
      float ps = p0 + p1;
#pragma unroll
      for (int off = 1; off < 16; off <<= 1) ps += __shfl_xor(ps, off, 16);
      rsum[r] = rsum[r] * f + ps;
      rmax[r] = nm;
      factor[r] = f;
      int m = r + 8 * lg;
      Pt[m * 40 + ln]      = (_Float16)p0;   // C-layout -> [m][k] in LDS
      Pt[m * 40 + 16 + ln] = (_Float16)p1;
    }
#pragma unroll
    for (int t = 0; t < 4; ++t)
#pragma unroll
      for (int r = 0; r < 8; ++r) o[t][r] *= factor[r];

    // single-wave workgroup: barrier is a NOP, order LDS explicitly
    asm volatile("s_wait_dscnt 0" ::: "memory");

    // O += P * V  (K=32 keys, N=64 d-cols -> 4 WMMAs)
    v16h pf;
    {
      const _Float16* pp = &Pt[ln * 40];
      v8h lo = *(const v8h*)(pp + lg * 8);
      v8h hi = *(const v8h*)(pp + 16 + lg * 8);
      pf = cat8(lo, hi);
    }
#pragma unroll
    for (int t = 0; t < 4; ++t) {
      const _Float16* vp = &Vt[(t * 16 + ln) * 40 + lg * 8];
      v8h lo = *(const v8h*)(vp);
      v8h hi = *(const v8h*)(vp + 16);
      o[t] = __builtin_amdgcn_wmma_f32_16x16x32_f16(false, pf, false, cat8(lo, hi),
                                                    (short)0, o[t], false, false);
    }
    asm volatile("s_wait_dscnt 0" ::: "memory");   // before next iter overwrites Pt/Vt
  }

  // normalize and store f16 [b*SEQ + qi][h*64 + d] for the projection GEMM
#pragma unroll
  for (int t = 0; t < 4; ++t)
#pragma unroll
    for (int r = 0; r < 8; ++r) {
      int qi = qbase + r + 8 * lg;
      if (qi >= SEQ) continue;
      float val = o[t][r] / rsum[r];
      long gm = (long)b * SEQ + qi;
      int  gn = h * HDIM + t * 16 + ln;
      AttOut[gm * EMBED + gn] = (_Float16)val;
    }
}

// ---------------- host launcher ----------------
extern "C" void kernel_launch(void* const* d_in, const int* in_sizes, int n_in,
                              void* d_out, int out_size, void* d_ws, size_t ws_size,
                              hipStream_t stream) {
  const float* x     = (const float*)d_in[0];
  const float* img   = (const float*)d_in[1];
  const float* amask = (const float*)d_in[2];
  const float* wqkv  = (const float*)d_in[3];
  const float* bqkv  = (const float*)d_in[4];
  const float* wproj = (const float*)d_in[5];
  const float* bproj = (const float*)d_in[6];
  const float* ukw   = (const float*)d_in[7];
  const float* ukb   = (const float*)d_in[8];
  const float* uvw   = (const float*)d_in[9];
  const float* uvb   = (const float*)d_in[10];

  // workspace carve-out (~50.4 MB total)
  char* ws = (char*)d_ws;
  size_t off = 0;
  auto carve = [&](size_t bytes) { char* p = ws + off; off += (bytes + 255) & ~(size_t)255; return p; };
  _Float16* Xh     = (_Float16*)carve((size_t)MPAD * EMBED * 2);
  _Float16* WqkvT  = (_Float16*)carve((size_t)NQKV * EMBED * 2);
  _Float16* WprojT = (_Float16*)carve((size_t)EMBED * EMBED * 2);
  _Float16* Qb     = (_Float16*)carve((size_t)BATCH * HEADS * 1024 * HDIM * 2);
  _Float16* Kb     = (_Float16*)carve((size_t)BATCH * HEADS * KEYS * HDIM * 2);
  _Float16* Vb     = (_Float16*)carve((size_t)BATCH * HEADS * KEYS * HDIM * 2);
  _Float16* AttOut = (_Float16*)carve((size_t)MPAD * EMBED * 2);
  (void)ws_size; (void)in_sizes; (void)n_in; (void)out_size;

  const int T = 256;
  cvt_x_kernel<<<(MPAD * EMBED + T - 1) / T, T, 0, stream>>>(x, Xh);
  transpose_f16_kernel<<<(NQKV * EMBED + T - 1) / T, T, 0, stream>>>(wqkv, WqkvT, EMBED, NQKV);
  transpose_f16_kernel<<<(EMBED * EMBED + T - 1) / T, T, 0, stream>>>(wproj, WprojT, EMBED, EMBED);
  img_kv_kernel<<<(2 * BATCH * EMBED + T - 1) / T, T, 0, stream>>>(img, ukw, ukb, uvw, uvb, Kb, Vb);

  dim3 gq(NQKV / 64, MPAD / 64);
  gemm_wmma_kernel<0><<<gq, 128, GEMM_LDS_BYTES, stream>>>(Xh, WqkvT, bqkv, EMBED, NQKV,
                                                           Qb, Kb, Vb, nullptr);

  attn_kernel<<<BATCH * HEADS * 64, 32, 0, stream>>>(Qb, Kb, Vb, amask, AttOut);

  dim3 gp(EMBED / 64, MPAD / 64);
  gemm_wmma_kernel<1><<<gp, 128, GEMM_LDS_BYTES, stream>>>(AttOut, WprojT, bproj, EMBED, EMBED,
                                                           nullptr, nullptr, nullptr, (float*)d_out);
}
